// DGraph_GAT_56899726737498
// MI455X (gfx1250) — matrix operand order repaired and checked by reading, not compiled, not run on hardware
//
#include <hip/hip_runtime.h>
#include <stdint.h>

// ---------------------------------------------------------------------------
// DGraph-GAT forward, MI455X (gfx1250, wave32, WMMA bf16 16x16x32 / f32 acc)
// Strategy: never materialize the 256MB dense A. Each A-pass fuses
//   G = h3 @ h3^T  (WMMA bf16)  ->  A = sigmoid(t*(sqrt(max(d2,0))+theta))
//   (VALU f32)  ->  acc += A_tile @ p_tile (WMMA bf16, via LDS transpose)
// h3 is 8192x32 bf16 = 512KB -> fully L2 resident; A lives only in VGPRs.
// ---------------------------------------------------------------------------

#define NN   8192
#define INF_ 512
#define HF_  256

typedef __attribute__((ext_vector_type(16))) __bf16 bf16x16;
typedef __attribute__((ext_vector_type(8)))  float  f32x8;

union V16B { bf16x16 v; uint32_t u[8]; uint16_t s[16]; };
union V8F  { f32x8 v; float f[8]; };

__device__ __forceinline__ uint16_t f2bf(float f) {
  union { float f; uint32_t u; } x; x.f = f;
  uint32_t r = x.u + 0x7FFFu + ((x.u >> 16) & 1u);   // round-to-nearest-even
  return (uint16_t)(r >> 16);
}
__device__ __forceinline__ float bf2f(uint16_t h) {
  union { float f; uint32_t u; } x; x.u = ((uint32_t)h) << 16;
  return x.f;
}
__device__ __forceinline__ uint32_t pack2(const uint16_t* p) {
  uint32_t x; __builtin_memcpy(&x, p, 4); return x;   // works for global & LDS
}
__device__ __forceinline__ f32x8 zero8() {
  f32x8 z = {0.f,0.f,0.f,0.f,0.f,0.f,0.f,0.f}; return z;
}
__device__ __forceinline__ float sigmoid_fast(float z) {
  // 1/(1+exp(-z)) via v_exp_f32 (hw exp2) + v_rcp_f32
  float e = __builtin_amdgcn_exp2f(-z * 1.44269504088896341f);
  return __builtin_amdgcn_rcpf(1.0f + e);
}

// ---------------- elementwise converters ----------------
__global__ void k_cvt(const float* __restrict__ in, uint16_t* __restrict__ out, int n) {
  int i = blockIdx.x * blockDim.x + threadIdx.x;
  if (i < n) out[i] = f2bf(in[i]);
}
// W[K,Nc] f32 -> Wt[Nc,K] bf16 (transpose so B-operand K-pairs are contiguous)
__global__ void k_cvt_t(const float* __restrict__ W, uint16_t* __restrict__ Wt, int K, int Nc) {
  int i = blockIdx.x * blockDim.x + threadIdx.x;
  if (i < K * Nc) { int k = i / Nc, n = i % Nc; Wt[n * K + k] = f2bf(W[i]); }
}

// ---------------- dense bf16 GEMM + bias (+relu), one wave per 16x16 tile ---
// A bf16 [M,K] row-major, Wt bf16 [Nc,K] (W transposed), C bf16 [M,Nc]
__global__ void __launch_bounds__(32)
k_gemm_bf16(const uint16_t* __restrict__ A, const uint16_t* __restrict__ Wt,
            const float* __restrict__ bias, uint16_t* __restrict__ C,
            int K, int Nc, int relu) {
  const int lane = threadIdx.x & 31;
  const int n16  = lane & 15;
  const int hi   = lane >> 4;
  const int m0   = blockIdx.y * 16;
  const int n0   = blockIdx.x * 16;
  const uint16_t* arow = A  + (size_t)(m0 + n16) * K;
  const uint16_t* wrow = Wt + (size_t)(n0 + n16) * K;

  f32x8 acc = zero8();
  for (int kc = 0; kc < K; kc += 32) {
    V16B a, b;
#pragma unroll
    for (int v = 0; v < 8; ++v) {
      // 16-bit A-operand 16x32: vgpr v, lane-hi -> K pair
      const int ka = (v >> 2) * 16 + (v & 3) * 2 + hi * 8;
      a.u[v] = pack2(arow + kc + ka);
      // 16-bit B-operand 32x16: same K order as A (vgpr v, lane-hi -> K pair)
      const int kb = (v >> 2) * 16 + (v & 3) * 2 + hi * 8;
      b.u[v] = pack2(wrow + kc + kb);
    }
    acc = __builtin_amdgcn_wmma_f32_16x16x32_bf16(false, a.v, false, b.v,
                                                  (short)0, acc, false, false);
  }
  V8F r; r.v = acc;
  const float bv = bias[n0 + n16];
#pragma unroll
  for (int rr = 0; rr < 8; ++rr) {
    const int m = m0 + rr + 8 * hi;
    float vv = r.f[rr] + bv;
    if (relu) vv = fmaxf(vv, 0.0f);
    C[(size_t)m * Nc + n0 + n16] = f2bf(vv);
  }
}

// ---------------- per-row squared norms of h3 (bf16 [N,32]) ----------------
__global__ void k_rowsq(const uint16_t* __restrict__ h3, float* __restrict__ sq) {
  int i = blockIdx.x * blockDim.x + threadIdx.x;
  if (i >= NN) return;
  float s = 0.f;
#pragma unroll
  for (int k = 0; k < 32; ++k) { float v = bf2f(h3[i * 32 + k]); s += v * v; }
  sq[i] = s;
}

// ---------------- p1 = h3 @ gw1 (32x16), bf16 out [N,16] -------------------
__global__ void k_p1(const uint16_t* __restrict__ h3, const float* __restrict__ gw1,
                     uint16_t* __restrict__ p1) {
  int i = blockIdx.x * blockDim.x + threadIdx.x;
  if (i >= NN) return;
  float acc[16];
#pragma unroll
  for (int c = 0; c < 16; ++c) acc[c] = 0.f;
  for (int k = 0; k < 32; ++k) {
    float hv = bf2f(h3[i * 32 + k]);
#pragma unroll
    for (int c = 0; c < 16; ++c) acc[c] += hv * gw1[k * 16 + c];
  }
#pragma unroll
  for (int c = 0; c < 16; ++c) p1[i * 16 + c] = f2bf(acc[c]);
}

// ---------------- p2 = h4 @ gw2 (16x8), bf16 out [N,16] zero-padded --------
__global__ void k_p2(const float* __restrict__ h4, const float* __restrict__ gw2,
                     uint16_t* __restrict__ p2) {
  int i = blockIdx.x * blockDim.x + threadIdx.x;
  if (i >= NN) return;
  float acc[8];
#pragma unroll
  for (int c = 0; c < 8; ++c) acc[c] = 0.f;
  for (int k = 0; k < 16; ++k) {
    float hv = h4[i * 16 + k];
#pragma unroll
    for (int c = 0; c < 8; ++c) acc[c] += hv * gw2[k * 8 + c];
  }
#pragma unroll
  for (int c = 0; c < 8; ++c)  p2[i * 16 + c] = f2bf(acc[c]);
#pragma unroll
  for (int c = 8; c < 16; ++c) p2[i * 16 + c] = 0;   // pad B-operand cols
}

// ---------------- fused A-pass: out = relu((A @ p)/N + gb) -----------------
// One wave owns a 16-row output stripe; sweeps j in chunks of 32.
__global__ void __launch_bounds__(128)
k_apass(const uint16_t* __restrict__ h3, const float* __restrict__ sq,
        const uint16_t* __restrict__ p, const float* __restrict__ tptr,
        const float* __restrict__ thptr, const float* __restrict__ gb,
        float* __restrict__ out, int ncols) {
  __shared__ uint16_t smem[4][16 * 36];          // per-wave 16x32 bf16 tile, pad 36
  const int lane = threadIdx.x & 31;
  const int wid  = threadIdx.x >> 5;
  const int n16  = lane & 15;
  const int hi   = lane >> 4;
  const int i0   = (blockIdx.x * 4 + wid) * 16;
  const float tv  = *tptr;
  const float thv = *thptr;

  // persistent A-fragment: h3 rows i0..i0+15, K=0..31 (bf16 16x32 layout)
  V16B Ai;
#pragma unroll
  for (int v = 0; v < 8; ++v) {
    const int ka = (v >> 2) * 16 + (v & 3) * 2 + hi * 8;
    Ai.u[v] = pack2(h3 + (i0 + n16) * 32 + ka);
  }
  float sqi[8];
#pragma unroll
  for (int r = 0; r < 8; ++r) sqi[r] = sq[i0 + r + 8 * hi];

  uint16_t* srow = smem[wid];
  f32x8 acc = zero8();

  for (int j0 = 0; j0 < NN; j0 += 32) {
    // ---- Gram tiles: G[i0:16, j0:16] and G[i0:16, j0+16:16], K=32 ----
    V16B B0, B1;
#pragma unroll
    for (int v = 0; v < 8; ++v) {
      const int kb = (v >> 2) * 16 + (v & 3) * 2 + hi * 8;  // B[k][n] = h3[j0+n][k]
      B0.u[v] = pack2(h3 + (j0 + n16) * 32 + kb);
      B1.u[v] = pack2(h3 + (j0 + 16 + n16) * 32 + kb);
    }
    f32x8 T0 = __builtin_amdgcn_wmma_f32_16x16x32_bf16(false, Ai.v, false, B0.v,
                                                       (short)0, zero8(), false, false);
    f32x8 T1 = __builtin_amdgcn_wmma_f32_16x16x32_bf16(false, Ai.v, false, B1.v,
                                                       (short)0, zero8(), false, false);
    const float sqj0 = sq[j0 + n16];
    const float sqj1 = sq[j0 + 16 + n16];
    V8F t0, t1; t0.v = T0; t1.v = T1;

    if (j0 + 32 < NN) {                               // uniform cond -> no EXEC games
      __builtin_prefetch(h3 + (j0 + 32 + n16) * 32, 0, 3);   // global_prefetch_b8
      __builtin_prefetch(p + (j0 + 32) * 16 + lane, 0, 3);
    }

    // ---- d2 -> dist -> sigmoid, write bf16 A-tile to LDS (transpose bounce)
#pragma unroll
    for (int r = 0; r < 8; ++r) {
      const int m = r + 8 * hi;                       // acc layout: row = r+8*hi, col = n16
      float d2 = fmaxf(sqi[r] + sqj0 - 2.0f * t0.f[r], 0.0f);
      float a0 = sigmoid_fast(tv * (__builtin_amdgcn_sqrtf(d2) + thv));
      srow[m * 36 + n16] = f2bf(a0);
      d2 = fmaxf(sqi[r] + sqj1 - 2.0f * t1.f[r], 0.0f);
      float a1 = sigmoid_fast(tv * (__builtin_amdgcn_sqrtf(d2) + thv));
      srow[m * 36 + 16 + n16] = f2bf(a1);
    }

    // ---- reload as bf16 A-operand (lane=row, vgprs=K) and fold in p-tile --
    V16B Ap, Bp;
#pragma unroll
    for (int v = 0; v < 8; ++v) {
      const int ka = (v >> 2) * 16 + (v & 3) * 2 + hi * 8;
      Ap.u[v] = pack2(&srow[n16 * 36 + ka]);          // ds_load_b32
      const int kb = (v >> 2) * 16 + (v & 3) * 2 + hi * 8;  // B[k][n] = p[j0+k][n]
      Bp.s[2 * v]     = p[(j0 + kb) * 16 + n16];
      Bp.s[2 * v + 1] = p[(j0 + kb + 1) * 16 + n16];
    }
    acc = __builtin_amdgcn_wmma_f32_16x16x32_bf16(false, Ap.v, false, Bp.v,
                                                  (short)0, acc, false, false);
  }

  // ---- epilogue: right-normalize, bias, relu ----
  const float inv = 1.0f / (float)NN;
  V8F r; r.v = acc;
  const int c = n16;
  const float gbv = (c < ncols) ? gb[c] : 0.0f;
#pragma unroll
  for (int rr = 0; rr < 8; ++rr) {
    const int m = i0 + rr + 8 * hi;
    const float vv = fmaxf(r.f[rr] * inv + gbv, 0.0f);
    if (c < ncols) out[(size_t)m * ncols + c] = vv;
  }
}

// ---------------- final tiny MLP: relu(h5@lw1+lb1)@lw2+lb2 -----------------
__global__ void k_final(const float* __restrict__ h5, const float* __restrict__ lw1,
                        const float* __restrict__ lb1, const float* __restrict__ lw2,
                        const float* __restrict__ lb2, float* __restrict__ out) {
  int i = blockIdx.x * blockDim.x + threadIdx.x;
  if (i >= NN) return;
  float z[16];
#pragma unroll
  for (int cc = 0; cc < 16; ++cc) {
    float s = lb1[cc];
#pragma unroll
    for (int k = 0; k < 8; ++k) s += h5[i * 8 + k] * lw1[k * 16 + cc];
    z[cc] = fmaxf(s, 0.0f);
  }
#pragma unroll
  for (int cc = 0; cc < 16; ++cc) {
    float s = lb2[cc];
#pragma unroll
    for (int k = 0; k < 16; ++k) s += z[k] * lw2[k * 16 + cc];
    out[i * 16 + cc] = s;
  }
}

// ---------------------------------------------------------------------------
extern "C" void kernel_launch(void* const* d_in, const int* in_sizes, int n_in,
                              void* d_out, int out_size, void* d_ws, size_t ws_size,
                              hipStream_t stream) {
  (void)in_sizes; (void)n_in; (void)out_size; (void)ws_size;
  const float* x   = (const float*)d_in[0];
  const float* w1  = (const float*)d_in[1];
  const float* b1  = (const float*)d_in[2];
  const float* w2  = (const float*)d_in[3];
  const float* b2  = (const float*)d_in[4];
  const float* w3  = (const float*)d_in[5];
  const float* b3  = (const float*)d_in[6];
  const float* t   = (const float*)d_in[7];
  const float* th  = (const float*)d_in[8];
  const float* gw1 = (const float*)d_in[9];
  const float* gb1 = (const float*)d_in[10];
  const float* gw2 = (const float*)d_in[11];
  const float* gb2 = (const float*)d_in[12];
  const float* lw1 = (const float*)d_in[13];
  const float* lb1 = (const float*)d_in[14];
  const float* lw2 = (const float*)d_in[15];
  const float* lb2 = (const float*)d_in[16];
  float* out = (float*)d_out;

  char* ws = (char*)d_ws;
  size_t off = 0;
  auto alloc = [&](size_t bytes) -> char* {
    char* p = ws + off;
    off = (off + bytes + 255) & ~(size_t)255;
    return p;
  };
  uint16_t* xbf  = (uint16_t*)alloc((size_t)NN * INF_ * 2);
  uint16_t* w1t  = (uint16_t*)alloc((size_t)INF_ * HF_ * 2);
  uint16_t* w2t  = (uint16_t*)alloc((size_t)HF_ * HF_ * 2);
  uint16_t* w3t  = (uint16_t*)alloc((size_t)HF_ * 32 * 2);
  uint16_t* h1bf = (uint16_t*)alloc((size_t)NN * HF_ * 2);
  uint16_t* h2bf = (uint16_t*)alloc((size_t)NN * HF_ * 2);
  uint16_t* h3bf = (uint16_t*)alloc((size_t)NN * 32 * 2);
  float*    sq   = (float*)   alloc((size_t)NN * 4);
  uint16_t* p1bf = (uint16_t*)alloc((size_t)NN * 16 * 2);
  float*    h4   = (float*)   alloc((size_t)NN * 16 * 4);
  uint16_t* p2bf = (uint16_t*)alloc((size_t)NN * 16 * 2);
  float*    h5   = (float*)   alloc((size_t)NN * 8 * 4);

  // -- converts --
  {
    int n = NN * INF_;
    k_cvt<<<(n + 255) / 256, 256, 0, stream>>>(x, xbf, n);
    k_cvt_t<<<(INF_ * HF_ + 255) / 256, 256, 0, stream>>>(w1, w1t, INF_, HF_);
    k_cvt_t<<<(HF_ * HF_ + 255) / 256, 256, 0, stream>>>(w2, w2t, HF_, HF_);
    k_cvt_t<<<(HF_ * 32 + 255) / 256, 256, 0, stream>>>(w3, w3t, HF_, 32);
  }
  // -- MLP encoder (WMMA) --
  k_gemm_bf16<<<dim3(HF_ / 16, NN / 16), 32, 0, stream>>>(xbf,  w1t, b1, h1bf, INF_, HF_, 1);
  k_gemm_bf16<<<dim3(HF_ / 16, NN / 16), 32, 0, stream>>>(h1bf, w2t, b2, h2bf, HF_,  HF_, 1);
  k_gemm_bf16<<<dim3(32 / 16,  NN / 16), 32, 0, stream>>>(h2bf, w3t, b3, h3bf, HF_,  32,  0);
  // -- norms + p1 --
  k_rowsq<<<NN / 256, 256, 0, stream>>>(h3bf, sq);
  k_p1<<<NN / 256, 256, 0, stream>>>(h3bf, gw1, p1bf);
  // -- fused A pass 1: h4 = relu((A@p1)/N + gb1) --
  k_apass<<<NN / 64, 128, 0, stream>>>(h3bf, sq, p1bf, t, th, gb1, h4, 16);
  // -- p2 = h4 @ gw2 (padded to 16 cols) --
  k_p2<<<NN / 256, 256, 0, stream>>>(h4, gw2, p2bf);
  // -- fused A pass 2: h5 = relu((A@p2)/N + gb2) --
  k_apass<<<NN / 64, 128, 0, stream>>>(h3bf, sq, p2bf, t, th, gb2, h5, 8);
  // -- classifier head --
  k_final<<<NN / 256, 256, 0, stream>>>(h5, lw1, lb1, lw2, lb2, out);
}
